// WFDSA_65120294142333
// MI455X (gfx1250) — matrix operand order repaired and checked by reading, not compiled, not run on hardware
//
#include <hip/hip_runtime.h>
#include <hip/hip_bf16.h>
#include <math.h>

// ---------------------------------------------------------------------------
// WFDSA (wavelet frequency-domain attention) for MI455X / gfx1250.
// All GEMM-like ops on v_wmma_f32_16x16x32_bf16 (wave32).
// K/V staging in attention uses GLOBAL_LOAD_ASYNC_TO_LDS_B128 when the
// toolchain exposes the gfx1250 async builtins (guarded, sync fallback).
// ---------------------------------------------------------------------------

typedef __attribute__((ext_vector_type(16))) __bf16 v16bf;
typedef __attribute__((ext_vector_type(8)))  float  v8f;
typedef __attribute__((ext_vector_type(4)))  int    v4i_t;

#define BATCH   2048
#define NTOK    256
#define CH      96
#define NHEAD   3
#define HDIM    32
#define NSUB    64      // 8x8 subband tokens
#define C3      288     // 3*CH

#if defined(__has_builtin)
#if __has_builtin(__builtin_amdgcn_global_load_async_to_lds_b128) && \
    __has_builtin(__builtin_amdgcn_s_wait_asynccnt)
#define WFDSA_ASYNC_LDS 1
#endif
#endif

__constant__ float U1c[16] = {
     0.8664f,  0.1026f,  0.4852f, -0.0574f,
    -0.1026f,  0.8664f, -0.0574f, -0.4852f,
     0.4852f,  0.0574f, -0.8664f,  0.1026f,
     0.0574f, -0.4852f, -0.1026f, -0.8664f};
__constant__ float U2c[16] = {
     1.3968f,  0.2212f, -0.2212f, -1.3968f,
    -0.2212f,  1.3968f, -1.3968f,  0.2212f,
    -0.5412f, -1.3066f, -1.3066f, -0.5412f,
     1.3066f, -0.5412f, -0.5412f,  1.3066f};

// ---- WMMA operand loaders (b128 fetches) ----------------------------------
// A operand (16x32 bf16): lane r=lane&15 -> row r; half h=lane>>4:
//   elem i -> K = k0 + h*16 + i
__device__ __forceinline__ v16bf load_A_row(const float* __restrict__ row,
                                            int k0, int hh) {
    const float4* p = (const float4*)(row + k0 + hh * 16);
    float4 q0 = p[0], q1 = p[1], q2 = p[2], q3 = p[3];
    v16bf a;
    a[0]  = (__bf16)q0.x; a[1]  = (__bf16)q0.y; a[2]  = (__bf16)q0.z; a[3]  = (__bf16)q0.w;
    a[4]  = (__bf16)q1.x; a[5]  = (__bf16)q1.y; a[6]  = (__bf16)q1.z; a[7]  = (__bf16)q1.w;
    a[8]  = (__bf16)q2.x; a[9]  = (__bf16)q2.y; a[10] = (__bf16)q2.z; a[11] = (__bf16)q2.w;
    a[12] = (__bf16)q3.x; a[13] = (__bf16)q3.y; a[14] = (__bf16)q3.z; a[15] = (__bf16)q3.w;
    return a;
}

// B operand for X @ W^T (W row-major [n][k]): lane n=lane&15, half h:
//   elem i -> K = k0 + h*16 + i   -> 16 contiguous floats of W row (col0+n)
__device__ __forceinline__ v16bf load_B_row16(const float* __restrict__ p) {
    const float4* q = (const float4*)p;
    float4 q0 = q[0], q1 = q[1], q2 = q[2], q3 = q[3];
    v16bf bm;
    bm[0]  = (__bf16)q0.x; bm[1]  = (__bf16)q0.y; bm[2]  = (__bf16)q0.z; bm[3]  = (__bf16)q0.w;
    bm[4]  = (__bf16)q1.x; bm[5]  = (__bf16)q1.y; bm[6]  = (__bf16)q1.z; bm[7]  = (__bf16)q1.w;
    bm[8]  = (__bf16)q2.x; bm[9]  = (__bf16)q2.y; bm[10] = (__bf16)q2.z; bm[11] = (__bf16)q2.w;
    bm[12] = (__bf16)q3.x; bm[13] = (__bf16)q3.y; bm[14] = (__bf16)q3.z; bm[15] = (__bf16)q3.w;
    return bm;
}

__device__ __forceinline__ v16bf load_B_wt(const float* __restrict__ W, int ldw,
                                           int col0, int k0, int lane) {
    int n  = lane & 15;
    int hh = lane >> 4;
    return load_B_row16(W + (size_t)(col0 + n) * ldw + k0 + hh * 16);
}

// ---- weight prep ----------------------------------------------------------
// w_t[(ky*3+kx)][n][ci] = conv_w[n][ci][ky][kx]
__global__ void prep_conv_w_kernel(const float* __restrict__ cw,
                                   float* __restrict__ w_t) {
    int i = blockIdx.x * blockDim.x + threadIdx.x;
    const int total = 9 * CH * C3;
    if (i >= total) return;
    int ci = i % C3;
    int n  = (i / C3) % CH;
    int kk = i / (C3 * CH);
    w_t[i] = cw[((size_t)n * C3 + ci) * 9 + kk];
}

// Wc = proj_w @ fuse_lh_w  (96 x 192),  bc = proj_w @ fuse_b + proj_b
__global__ void prep_fuse_kernel(const float* __restrict__ proj_w,
                                 const float* __restrict__ proj_b,
                                 const float* __restrict__ fuse_w,
                                 const float* __restrict__ fuse_b,
                                 float* __restrict__ Wc, float* __restrict__ bc) {
    int i = blockIdx.x * blockDim.x + threadIdx.x;
    if (i < CH * 2 * CH) {
        int o = i / (2 * CH), j = i % (2 * CH);
        float s = 0.f;
        for (int c = 0; c < CH; ++c) s += proj_w[o * CH + c] * fuse_w[c * 2 * CH + j];
        Wc[i] = s;
    }
    if (i < CH) {
        float s = proj_b[i];
        for (int c = 0; c < CH; ++c) s += proj_w[i * CH + c] * fuse_b[c];
        bc[i] = s;
    }
}

// ---- NLWT: A subband -> xl0 (B,64,96); B,C,D -> high_t (B,64,288) ---------
__global__ void nlwt_kernel(const float* __restrict__ x,
                            float* __restrict__ xl0,
                            float* __restrict__ high_t) {
    size_t idx = (size_t)blockIdx.x * blockDim.x + threadIdx.x;
    const size_t total = (size_t)BATCH * NSUB * CH;
    if (idx >= total) return;
    int c = (int)(idx % CH);
    int p = (int)((idx / CH) % NSUB);
    size_t b = idx / ((size_t)CH * NSUB);
    int y = p >> 3, xq = p & 7;
    const float* xb = x + b * NTOK * CH;

    auto x1row = [&](int yy, int xx2, int i) -> float {
        float s = 0.f;
#pragma unroll
        for (int dy = 0; dy < 2; ++dy)
#pragma unroll
            for (int dx = 0; dx < 2; ++dx)
                s += U1c[i * 4 + dy * 2 + dx] *
                     xb[((2 * yy + dy) * 16 + (2 * xx2 + dx)) * CH + c];
        return s;
    };
    int y1  = (y + 1) & 7;
    int x1i = (xq + 1) & 7;
    float s0 = x1row(y,  xq,  0);   // t[0] at (y,x)
    float s1 = x1row(y1, xq,  1);   // roll -1 along y
    float s2 = x1row(y,  x1i, 2);   // roll -1 along x
    float s3 = x1row(y1, x1i, 3);   // roll both

    float r0 = U2c[0]  * s0 + U2c[1]  * s1 + U2c[2]  * s2 + U2c[3]  * s3;
    float r1 = U2c[4]  * s0 + U2c[5]  * s1 + U2c[6]  * s2 + U2c[7]  * s3;
    float r2 = U2c[8]  * s0 + U2c[9]  * s1 + U2c[10] * s2 + U2c[11] * s3;
    float r3 = U2c[12] * s0 + U2c[13] * s1 + U2c[14] * s2 + U2c[15] * s3;

    size_t pb = b * NSUB + p;
    xl0[pb * CH + c] = r0;
    high_t[pb * C3 + c]           = r1;
    high_t[pb * C3 + CH + c]      = r2;
    high_t[pb * C3 + 2 * CH + c]  = r3;
}

// ---- generic GEMM: C(MxN) = A(MxK) @ W(NxK)^T + bias ----------------------
__global__ __launch_bounds__(128)
void gemm_xwt_bias(const float* __restrict__ A, const float* __restrict__ W,
                   const float* __restrict__ bias, float* __restrict__ C,
                   int M, int N, int K) {
    int tiles_n = N >> 4;
    int tiles_m = M >> 4;
    int tile = blockIdx.x * 4 + (threadIdx.x >> 5);
    if (tile >= tiles_m * tiles_n) return;
    int tm = tile / tiles_n, tn = tile % tiles_n;
    int row0 = tm << 4, col0 = tn << 4;
    int lane = threadIdx.x & 31;
    int r = lane & 15, hh = lane >> 4;
    const float* arow = A + (size_t)(row0 + r) * K;

    v8f acc = {};
    for (int k0 = 0; k0 < K; k0 += 32) {
        __builtin_prefetch(arow + k0 + 32 + hh * 8, 0, 0);
        v16bf a  = load_A_row(arow, k0, hh);
        v16bf bm = load_B_wt(W, K, col0, k0, lane);
        acc = __builtin_amdgcn_wmma_f32_16x16x32_bf16(
                  false, a, false, bm, (short)0, acc, false, false);
    }
    int cc = col0 + (lane & 15);
    float bv = bias[cc];
#pragma unroll
    for (int j = 0; j < 8; ++j) {
        int rr = row0 + j + hh * 8;
        C[(size_t)rr * N + cc] = acc[j] + bv;
    }
}

// ---- 3x3 conv over (B, 288, 8, 8) as implicit GEMM + LeakyReLU(0.2) -------
// input high_t is pixel-major (B*64, 288); w_t is (tap, n, ci)
__global__ __launch_bounds__(128)
void conv3x3_kernel(const float* __restrict__ high_t,
                    const float* __restrict__ w_t,
                    const float* __restrict__ bias,
                    float* __restrict__ outp) {
    const int M = BATCH * NSUB;
    const int tiles_n = CH >> 4;   // 6
    int tile = blockIdx.x * 4 + (threadIdx.x >> 5);
    if (tile >= (M >> 4) * tiles_n) return;
    int tm = tile / tiles_n, tn = tile % tiles_n;
    int row0 = tm << 4, col0 = tn << 4;
    int lane = threadIdx.x & 31;
    int r = lane & 15, hh = lane >> 4;
    int row = row0 + r;
    int b = row >> 6, p = row & 63, y = p >> 3, xx = p & 7;

    v8f acc = {};
#pragma unroll
    for (int ky = 0; ky < 3; ++ky)
#pragma unroll
        for (int kx = 0; kx < 3; ++kx) {
            int yy = y + ky - 1, xs = xx + kx - 1;
            bool ok = ((unsigned)yy < 8u) && ((unsigned)xs < 8u);
            const float* arow = high_t + ((size_t)b * NSUB + yy * 8 + xs) * C3;
            const float* wmat = w_t + (size_t)(ky * 3 + kx) * CH * C3;
            for (int k0 = 0; k0 < C3; k0 += 32) {
                v16bf a;
                if (ok) {
                    a = load_A_row(arow, k0, hh);
                } else {
#pragma unroll
                    for (int i = 0; i < 16; ++i) a[i] = (__bf16)0.f;
                }
                v16bf bm = load_B_wt(wmat, C3, col0, k0, lane);
                acc = __builtin_amdgcn_wmma_f32_16x16x32_bf16(
                          false, a, false, bm, (short)0, acc, false, false);
            }
        }
    int cc = col0 + (lane & 15);
    float bv = bias[cc];
#pragma unroll
    for (int j = 0; j < 8; ++j) {
        int rr = row0 + j + hh * 8;
        float v = acc[j] + bv;
        v = v >= 0.f ? v : 0.2f * v;
        outp[(size_t)rr * CH + cc] = v;
    }
}

// ---- cosine attention, one block per (b, head); 8 waves -------------------
// q: (B*256, 96); kv: (B*64, 192) [k at h*32, v at 96+h*32]
// out: attn_cat (B*256, 192), channel = dd*3 + h + chan_off
__global__ __launch_bounds__(256)
void attention_kernel(const float* __restrict__ qout,
                      const float* __restrict__ kv,
                      const float* __restrict__ ls_ptr,
                      float* __restrict__ attn_cat, int chan_off) {
    __shared__ float kn_s[NSUB * HDIM];   // normalized K rows
    __shared__ float v_s[NSUB * HDIM];    // V rows
    __shared__ float qinv[NTOK];
    __shared__ float Pbuf[8][16 * 64];    // per-wave P staging

    int b = blockIdx.x / NHEAD;
    int h = blockIdx.x % NHEAD;
    int tid = threadIdx.x;

#ifdef WFDSA_ASYNC_LDS
    // gfx1250 async DMA: 1024 b128 chunks (64 rows x (8 K + 8 V) chunks),
    // 4 per thread, tracked on ASYNCcnt.
    {
#pragma unroll
        for (int i = 0; i < 4; ++i) {
            int cid = (tid << 2) | i;
            int row = cid >> 4;
            int sub = cid & 15;
            int isv = sub >> 3;
            int ch  = sub & 7;
            const float* g = kv + ((size_t)b * NSUB + row) * 192 + h * HDIM
                               + isv * CH + (ch << 2);
            float* l = (isv ? v_s : kn_s) + row * HDIM + (ch << 2);
            __builtin_amdgcn_global_load_async_to_lds_b128(
                (__attribute__((address_space(1))) v4i_t*)(void*)g,
                (__attribute__((address_space(3))) v4i_t*)(void*)l, 0, 0);
        }
        __builtin_amdgcn_s_wait_asynccnt(0);
    }
    __syncthreads();
    if (tid < NSUB) {   // normalize K rows in place
        float ss = 0.f;
#pragma unroll
        for (int j = 0; j < HDIM; ++j) { float t = kn_s[tid * HDIM + j]; ss += t * t; }
        float inv = 1.f / fmaxf(sqrtf(ss), 1e-12f);
#pragma unroll
        for (int j = 0; j < HDIM; ++j) kn_s[tid * HDIM + j] *= inv;
    }
#else
    if (tid < NSUB) {
        const float* krow = kv + ((size_t)b * NSUB + tid) * 192 + h * HDIM;
        float ss = 0.f;
        float tmp[HDIM];
#pragma unroll
        for (int j = 0; j < HDIM; ++j) { tmp[j] = krow[j]; ss += tmp[j] * tmp[j]; }
        float inv = 1.f / fmaxf(sqrtf(ss), 1e-12f);
#pragma unroll
        for (int j = 0; j < HDIM; ++j) kn_s[tid * HDIM + j] = tmp[j] * inv;
        const float* vrow = krow + CH;
#pragma unroll
        for (int j = 0; j < HDIM; ++j) v_s[tid * HDIM + j] = vrow[j];
    }
#endif
    {
        const float* qrow = qout + ((size_t)b * NTOK + tid) * CH + h * HDIM;
        float ss = 0.f;
#pragma unroll
        for (int j = 0; j < HDIM; ++j) { float t = qrow[j]; ss += t * t; }
        qinv[tid] = 1.f / fmaxf(sqrtf(ss), 1e-12f);
    }
    __syncthreads();

    float scale = __expf(fminf(ls_ptr[h], 4.6051701859880914f)); // log(100)

    int wv = tid >> 5;
    int lane = tid & 31;
    int r = lane & 15, hh = lane >> 4;

    for (int tblk = 0; tblk < 2; ++tblk) {
        int row0 = (wv * 2 + tblk) << 4;
        int n = row0 + r;
        const float* qrow = qout + ((size_t)b * NTOK + n) * CH + h * HDIM;
        float qi = qinv[n];
        v16bf a;
        {
            const float4* p = (const float4*)(qrow + hh * 16);
            float4 q0 = p[0], q1 = p[1], q2 = p[2], q3 = p[3];
            a[0]  = (__bf16)(q0.x * qi); a[1]  = (__bf16)(q0.y * qi);
            a[2]  = (__bf16)(q0.z * qi); a[3]  = (__bf16)(q0.w * qi);
            a[4]  = (__bf16)(q1.x * qi); a[5]  = (__bf16)(q1.y * qi);
            a[6]  = (__bf16)(q1.z * qi); a[7]  = (__bf16)(q1.w * qi);
            a[8]  = (__bf16)(q2.x * qi); a[9]  = (__bf16)(q2.y * qi);
            a[10] = (__bf16)(q2.z * qi); a[11] = (__bf16)(q2.w * qi);
            a[12] = (__bf16)(q3.x * qi); a[13] = (__bf16)(q3.y * qi);
            a[14] = (__bf16)(q3.z * qi); a[15] = (__bf16)(q3.w * qi);
        }

        // S = qn @ kn^T over 64 keys: 4 tiles, K=32 each
        float s[4][8];
#pragma unroll
        for (int t = 0; t < 4; ++t) {
            v16bf bm = load_B_row16(&kn_s[(t * 16 + (lane & 15)) * HDIM + hh * 16]);
            v8f c = {};
            c = __builtin_amdgcn_wmma_f32_16x16x32_bf16(
                    false, a, false, bm, (short)0, c, false, false);
#pragma unroll
            for (int j = 0; j < 8; ++j) s[t][j] = c[j] * scale;
        }

        // softmax over 64 keys (row reduce: 4 local tiles + 16-lane halves)
#pragma unroll
        for (int j = 0; j < 8; ++j) {
            float m = fmaxf(fmaxf(s[0][j], s[1][j]), fmaxf(s[2][j], s[3][j]));
            for (int d = 1; d < 16; d <<= 1) m = fmaxf(m, __shfl_xor(m, d, 32));
            float p[4], sum = 0.f;
#pragma unroll
            for (int t = 0; t < 4; ++t) { p[t] = __expf(s[t][j] - m); sum += p[t]; }
            for (int d = 1; d < 16; d <<= 1) sum += __shfl_xor(sum, d, 32);
            float inv = 1.f / sum;
#pragma unroll
            for (int t = 0; t < 4; ++t)
                Pbuf[wv][(j + hh * 8) * 64 + t * 16 + (lane & 15)] = p[t] * inv;
        }

        // O = P @ V : two 16-col output tiles, K=64 (2 chunks)
#pragma unroll
        for (int dt = 0; dt < 2; ++dt) {
            v8f acc = {};
#pragma unroll
            for (int kc = 0; kc < 2; ++kc) {
                v16bf pa = load_A_row(&Pbuf[wv][r * 64], kc * 32, hh);
                v16bf vb;
#pragma unroll
                for (int i = 0; i < 16; ++i)
                    vb[i] = (__bf16)v_s[(kc * 32 + hh * 16 + i) * HDIM + dt * 16 + (lane & 15)];
                acc = __builtin_amdgcn_wmma_f32_16x16x32_bf16(
                          false, pa, false, vb, (short)0, acc, false, false);
            }
#pragma unroll
            for (int j = 0; j < 8; ++j) {
                int n2 = row0 + j + hh * 8;
                int dd = dt * 16 + (lane & 15);
                attn_cat[((size_t)b * NTOK + n2) * 192 + dd * NHEAD + h + chan_off] = acc[j];
            }
        }
    }
}

// ---------------------------------------------------------------------------
extern "C" void kernel_launch(void* const* d_in, const int* in_sizes, int n_in,
                              void* d_out, int out_size, void* d_ws, size_t ws_size,
                              hipStream_t stream) {
    (void)in_sizes; (void)n_in; (void)out_size; (void)ws_size;
    const float* x         = (const float*)d_in[0];
    const float* q_w       = (const float*)d_in[1];
    const float* q_b       = (const float*)d_in[2];
    const float* kv_low_w  = (const float*)d_in[3];
    const float* kv_low_b  = (const float*)d_in[4];
    const float* kv_high_w = (const float*)d_in[5];
    const float* kv_high_b = (const float*)d_in[6];
    const float* conv_w    = (const float*)d_in[7];
    const float* conv_b    = (const float*)d_in[8];
    const float* fuse_w    = (const float*)d_in[9];
    const float* fuse_b    = (const float*)d_in[10];
    const float* proj_w    = (const float*)d_in[11];
    const float* proj_b    = (const float*)d_in[12];
    const float* ls_low    = (const float*)d_in[13];
    const float* ls_high   = (const float*)d_in[14];
    float* out = (float*)d_out;

    float* ws = (float*)d_ws;
    size_t off = 0;
    const size_t Mq  = (size_t)BATCH * NTOK;   // 524288
    const size_t Ms  = (size_t)BATCH * NSUB;   // 131072
    float* q_out  = ws + off; off += Mq * CH;
    float* xl0    = ws + off; off += Ms * CH;
    float* high_t = ws + off; off += Ms * C3;
    float* kvl    = ws + off; off += Ms * 2 * CH;
    float* convo  = ws + off; off += Ms * CH;
    float* kvh    = ws + off; off += Ms * 2 * CH;
    float* cat    = ws + off; off += Mq * 2 * CH;
    float* w_t    = ws + off; off += (size_t)9 * CH * C3;
    float* Wc     = ws + off; off += (size_t)CH * 2 * CH;
    float* bc     = ws + off; off += CH;

    // weight prep
    prep_conv_w_kernel<<<(9 * CH * C3 + 255) / 256, 256, 0, stream>>>(conv_w, w_t);
    prep_fuse_kernel<<<(CH * 2 * CH + 255) / 256, 256, 0, stream>>>(
        proj_w, proj_b, fuse_w, fuse_b, Wc, bc);

    // wavelet transform
    {
        size_t total = (size_t)BATCH * NSUB * CH;
        nlwt_kernel<<<(unsigned)((total + 255) / 256), 256, 0, stream>>>(x, xl0, high_t);
    }

    // q projection: (Mq x 96) = x @ q_w^T + q_b
    {
        int tiles = (int)(Mq / 16) * (CH / 16);
        gemm_xwt_bias<<<(tiles + 3) / 4, 128, 0, stream>>>(x, q_w, q_b, q_out,
                                                           (int)Mq, CH, CH);
    }
    // kv_low: (Ms x 192) = xl0 @ kv_low_w^T + b
    {
        int tiles = (int)(Ms / 16) * (2 * CH / 16);
        gemm_xwt_bias<<<(tiles + 3) / 4, 128, 0, stream>>>(xl0, kv_low_w, kv_low_b,
                                                           kvl, (int)Ms, 2 * CH, CH);
    }
    // high branch conv (implicit GEMM) + leaky relu
    {
        int tiles = (int)(Ms / 16) * (CH / 16);
        conv3x3_kernel<<<(tiles + 3) / 4, 128, 0, stream>>>(high_t, w_t, conv_b, convo);
    }
    // kv_high
    {
        int tiles = (int)(Ms / 16) * (2 * CH / 16);
        gemm_xwt_bias<<<(tiles + 3) / 4, 128, 0, stream>>>(convo, kv_high_w, kv_high_b,
                                                           kvh, (int)Ms, 2 * CH, CH);
    }
    // attention, both branches
    attention_kernel<<<BATCH * NHEAD, 256, 0, stream>>>(q_out, kvl, ls_low, cat, 0);
    attention_kernel<<<BATCH * NHEAD, 256, 0, stream>>>(q_out, kvh, ls_high, cat, CH);

    // fused fuse_lh + proj: out = cat @ Wc^T + bc
    {
        int tiles = (int)(Mq / 16) * (CH / 16);
        gemm_xwt_bias<<<(tiles + 3) / 4, 128, 0, stream>>>(cat, Wc, bc, out,
                                                           (int)Mq, CH, 2 * CH);
    }
}
